// TimestepEmbedder_8254927143046
// MI455X (gfx1250) — matrix operand order repaired and checked by reading, not compiled, not run on hardware
//
#include <hip/hip_runtime.h>

typedef __attribute__((ext_vector_type(16))) _Float16 v16h;
typedef __attribute__((ext_vector_type(8)))  _Float16 v8h;
typedef __attribute__((ext_vector_type(8)))  float    v8f;
typedef __attribute__((ext_vector_type(4)))  float    v4f;

#define BATCH   32
#define HIDDEN  1152
#define FREQ    256
#define SEQ     4096

// ---------------------------------------------------------------------------
// Kernel 1: sinusoidal timestep embedding [32,256] -> f16 (row-major)
// ---------------------------------------------------------------------------
__global__ void emb_kernel(const float* __restrict__ t, _Float16* __restrict__ emb) {
    const int b = blockIdx.x;        // 0..31
    const int f = threadIdx.x;       // 0..255
    const int half = FREQ / 2;
    const float tv = t[b];
    const int j = (f < half) ? f : (f - half);
    const float freq = __expf(-logf(10000.0f) * (float)j / (float)half);
    const float arg  = tv * freq;
    const float v    = (f < half) ? cosf(arg) : sinf(arg);
    emb[b * FREQ + f] = (_Float16)v;
}

// ---------------------------------------------------------------------------
// WMMA fragment loaders (CDNA5 ISA 7.12.2 layouts, wave32)
// ---------------------------------------------------------------------------
// A 16x32 f16 (MxK): lane l -> M = l%16; elems 0..7 = K0+(l>=16?8:0)+0..7,
//                                 elems 8..15 = K0+16+(l>=16?8:0)+0..7
__device__ __forceinline__
v16h load_a_frag(const _Float16* __restrict__ A, int ldk, int M, int k0, int lane) {
    const _Float16* row = A + (size_t)M * ldk;
    const int kb = k0 + ((lane >= 16) ? 8 : 0);
    v8h lo = *(const v8h*)(row + kb);
    v8h hi = *(const v8h*)(row + kb + 16);
    v16h r;
#pragma unroll
    for (int i = 0; i < 8; ++i) { r[i] = lo[i]; r[8 + i] = hi[i]; }
    return r;
}

// B 32x16 f16 (KxN): lane l -> N = l%16; elems 0..7 = K0+(l/16)*8+0..7, elems 8..15 = K0+16+(l/16)*8+0..7.
// Source weight W[N][K] is f32 row-major; two runs of 8 consecutive f32 (at +0 and +16), cvt to f16.
__device__ __forceinline__
v16h load_b_frag_f32(const float* __restrict__ W, int ldk, int N, int k0, int lane) {
    const float* row = W + (size_t)N * ldk + k0 + ((lane >= 16) ? 8 : 0);
    v16h r;
#pragma unroll
    for (int i = 0; i < 4; ++i) {
        v4f q = *(const v4f*)(row + (i & 1) * 4 + (i >> 1) * 16);
        r[i * 4 + 0] = (_Float16)q[0];
        r[i * 4 + 1] = (_Float16)q[1];
        r[i * 4 + 2] = (_Float16)q[2];
        r[i * 4 + 3] = (_Float16)q[3];
    }
    return r;
}

__device__ __forceinline__ float gelu_tanh(float x) {
    const float k0 = 0.7978845608028654f;   // sqrt(2/pi)
    const float k1 = 0.044715f;
    return 0.5f * x * (1.0f + tanhf(k0 * (x + k1 * x * x * x)));
}

// ---------------------------------------------------------------------------
// Kernel 2: h = gelu(emb @ W1^T + b1)  -> f16 [32,1152]
// grid = 2 M-tiles * 72 N-tiles, block = 32 (one wave, EXEC all-1s for WMMA)
// ---------------------------------------------------------------------------
__global__ void gemm1_kernel(const _Float16* __restrict__ emb,
                             const float*   __restrict__ W1,
                             const float*   __restrict__ b1,
                             _Float16*      __restrict__ h) {
    const int lane = threadIdx.x;
    const int mt = blockIdx.x & 1;
    const int n0 = (blockIdx.x >> 1) * 16;
    v8f acc = {};
#pragma unroll
    for (int k0 = 0; k0 < FREQ; k0 += 32) {
        v16h a = load_a_frag(emb, FREQ, mt * 16 + (lane & 15), k0, lane);
        v16h b = load_b_frag_f32(W1, FREQ, n0 + (lane & 15), k0, lane);
        acc = __builtin_amdgcn_wmma_f32_16x16x32_f16(false, a, false, b,
                                                     (short)0, acc, false, false);
    }
    const int N = n0 + (lane & 15);
    const float bias = b1[N];
#pragma unroll
    for (int r = 0; r < 8; ++r) {
        const int M = mt * 16 + r + ((lane >= 16) ? 8 : 0);
        h[(size_t)M * HIDDEN + N] = (_Float16)gelu_tanh(acc[r] + bias);
    }
}

// ---------------------------------------------------------------------------
// Kernel 3: out32 = h @ W2^T + b2  -> f32 [32,1152]
// ---------------------------------------------------------------------------
__global__ void gemm2_kernel(const _Float16* __restrict__ h,
                             const float*   __restrict__ W2,
                             const float*   __restrict__ b2,
                             float*         __restrict__ out32) {
    const int lane = threadIdx.x;
    const int mt = blockIdx.x & 1;
    const int n0 = (blockIdx.x >> 1) * 16;
    v8f acc = {};
    for (int k0 = 0; k0 < HIDDEN; k0 += 32) {
        v16h a = load_a_frag(h, HIDDEN, mt * 16 + (lane & 15), k0, lane);
        v16h b = load_b_frag_f32(W2, HIDDEN, n0 + (lane & 15), k0, lane);
        acc = __builtin_amdgcn_wmma_f32_16x16x32_f16(false, a, false, b,
                                                     (short)0, acc, false, false);
    }
    const int N = n0 + (lane & 15);
    const float bias = b2[N];
#pragma unroll
    for (int r = 0; r < 8; ++r) {
        const int M = mt * 16 + r + ((lane >= 16) ? 8 : 0);
        out32[(size_t)M * HIDDEN + N] = acc[r] + bias;
    }
}

// ---------------------------------------------------------------------------
// Kernel 4: broadcast [32,1152] -> [32,1152,1,4096]. 604 MB streaming stores;
// this is THE bandwidth-bound phase. One block per contiguous 16 KB run,
// 128-bit non-temporal stores (write-once data: bypass/rinse caches).
// ---------------------------------------------------------------------------
__global__ void bcast_kernel(const float* __restrict__ src, float* __restrict__ dst) {
    const int pair = blockIdx.x;                 // b*HIDDEN + h
    const float v  = src[pair];                  // uniform -> scalar load
    v4f fv = {v, v, v, v};
    v4f* base = (v4f*)(dst + (size_t)pair * SEQ);
    const int tid = threadIdx.x;                 // 0..255
#pragma unroll
    for (int i = 0; i < 4; ++i) {
        __builtin_nontemporal_store(fv, base + i * 256 + tid);
    }
}

// ---------------------------------------------------------------------------
extern "C" void kernel_launch(void* const* d_in, const int* in_sizes, int n_in,
                              void* d_out, int out_size, void* d_ws, size_t ws_size,
                              hipStream_t stream) {
    const float* t  = (const float*)d_in[0];   // [32]
    const float* W1 = (const float*)d_in[1];   // [1152,256]
    const float* b1 = (const float*)d_in[2];   // [1152]
    const float* W2 = (const float*)d_in[3];   // [1152,1152]
    const float* b2 = (const float*)d_in[4];   // [1152]
    float* out = (float*)d_out;                // [32,1152,1,4096]

    char* ws = (char*)d_ws;
    _Float16* emb   = (_Float16*)ws;                          // 16384 B
    _Float16* h     = (_Float16*)(ws + 16384);                // 73728 B
    float*    out32 = (float*)(ws + 16384 + 73728);           // 147456 B

    emb_kernel  <<<BATCH, FREQ, 0, stream>>>(t, emb);
    gemm1_kernel<<<2 * (HIDDEN / 16), 32, 0, stream>>>(emb, W1, b1, h);
    gemm2_kernel<<<2 * (HIDDEN / 16), 32, 0, stream>>>(h, W2, b2, out32);
    bcast_kernel<<<BATCH * HIDDEN, 256, 0, stream>>>(out32, out);
}
